// GINmolbbbp_82815559401961
// MI455X (gfx1250) — matrix operand order repaired and checked by reading, not compiled, not run on hardware
//
#include <hip/hip_runtime.h>
#include <hip/hip_bf16.h>

#define NN 50000
#define EE 800000
#define CC 128
#define EDD 16
#define GG 512
#define EPB 64

typedef __attribute__((ext_vector_type(16))) _Float16 v16h;
typedef __attribute__((ext_vector_type(8)))  _Float16 v8h;
typedef __attribute__((ext_vector_type(8)))  float    v8f;

// ---------------- utility ----------------
__global__ __launch_bounds__(256) void zero_f32(float* __restrict__ p, long n) {
    long i = (long)blockIdx.x * 256 + threadIdx.x;
    if (i < n) p[i] = 0.0f;
}

__global__ __launch_bounds__(256) void wconv_f16(const float* __restrict__ s,
                                                 _Float16* __restrict__ d, int n) {
    int i = blockIdx.x * 256 + threadIdx.x;
    if (i < n) d[i] = (_Float16)s[i];
}

// ---------------- edge message + scatter-add ----------------
// msg = relu(x[src] + edge_attr @ lw^T + lb); agg[dst] += msg
__global__ __launch_bounds__(256) void edge_msg(
    const float* __restrict__ xin, const int* __restrict__ srci,
    const int* __restrict__ dsti, const float* __restrict__ ea,
    const float* __restrict__ lw, const float* __restrict__ lb,
    float* __restrict__ agg)
{
    __shared__ float lwsh[CC * 17];
    __shared__ float lbsh[CC];
    for (int i = threadIdx.x; i < CC * EDD; i += 256) {
        int c = i >> 4, k = i & 15;
        lwsh[c * 17 + k] = lw[i];
    }
    if (threadIdx.x < CC) lbsh[threadIdx.x] = lb[threadIdx.x];
    __syncthreads();

    const int c    = threadIdx.x & 127;
    const int slot = threadIdx.x >> 7;
    const long e0  = (long)blockIdx.x * EPB;

    for (int it = 0; it < EPB / 2; ++it) {
        long e = e0 + it * 2 + slot;
        if (e >= EE) break;
        // prefetch the x-row of the edge two ahead (global_prefetch_b8)
        if (e + 2 < EE) {
            int sp = srci[e + 2];
            __builtin_prefetch(xin + (long)sp * CC + c, 0, 1);
        }
        int s = srci[e];
        int d = dsti[e];
        const float4* ea4 = (const float4*)(ea + e * EDD);   // 64B-aligned rows
        float4 q0 = ea4[0], q1 = ea4[1], q2 = ea4[2], q3 = ea4[3];
        const float att[16] = {q0.x,q0.y,q0.z,q0.w, q1.x,q1.y,q1.z,q1.w,
                               q2.x,q2.y,q2.z,q2.w, q3.x,q3.y,q3.z,q3.w};
        float dot = 0.0f;
        #pragma unroll
        for (int k = 0; k < EDD; ++k) dot += att[k] * lwsh[c * 17 + k];
        float v = xin[(long)s * CC + c] + dot + lbsh[c];
        v = v > 0.0f ? v : 0.0f;
        atomicAdd(&agg[(long)d * CC + c], v);
    }
}

// ---------------- WMMA GEMM: out[N,128] = f(A)[N,128] @ W^T + bias ----------------
// MODE 0: a(m,k) = A0[m,k] + (1+eps)*A1[m,k];           epilogue: acc + bias
// MODE 1: a(m,k) = relu(A0[m,k]*bnp[k] + bnp[128+k]);   epilogue: relu(acc + bias)
// Block = 256 threads = 8 wave32 waves; wave owns a 16-row x 128-col strip.
// Weights staged in LDS with padded row stride (136 f16 = 68 dwords -> bank 4n per lane).
template<int MODE>
__global__ __launch_bounds__(256) void gemm128(
    const float* __restrict__ A0, const float* __restrict__ A1,
    const float* __restrict__ epsp, const float* __restrict__ bnp,
    const _Float16* __restrict__ Wh, const float* __restrict__ bias,
    float* __restrict__ out)
{
    __shared__ __align__(16) _Float16 wsh[CC * 136];
    // stage 128x128 f16 weights: each of 2048 chunks is 8 contiguous f16 (16B)
    for (int i = threadIdx.x; i < CC * 16; i += 256) {
        int row = i >> 4, seg = i & 15;
        *(float4*)(wsh + row * 136 + seg * 8) = *(const float4*)(Wh + row * CC + seg * 8);
    }
    __syncthreads();

    const int lane = threadIdx.x & 31;
    const int wave = threadIdx.x >> 5;
    const int half = lane >> 4;
    const int l16  = lane & 15;
    const long rowBase = (long)blockIdx.x * 128 + wave * 16;

    float coef = 0.0f;
    if (MODE == 0) coef = 1.0f + epsp[0];

    v8f acc[8];
    #pragma unroll
    for (int i = 0; i < 8; ++i) acc[i] = (v8f){0,0,0,0,0,0,0,0};

    long m = rowBase + l16;
    if (m >= NN) m = NN - 1;                 // clamped rows never stored
    const float* a0row = A0 + m * CC;
    const float* a1row = (MODE == 0) ? (A1 + m * CC) : nullptr;

    #pragma unroll
    for (int kt = 0; kt < 4; ++kt) {
        const int base = kt * 32 + half * 8; // group A: k=[base,base+8); group B: +16
        // ---- vectorized A loads (rows are 512B, base multiple of 8 -> 32B aligned) ----
        float4 a0 = *(const float4*)(a0row + base);
        float4 a1 = *(const float4*)(a0row + base + 4);
        float4 b0 = *(const float4*)(a0row + base + 16);
        float4 b1 = *(const float4*)(a0row + base + 20);
        float ga[8] = {a0.x,a0.y,a0.z,a0.w, a1.x,a1.y,a1.z,a1.w};
        float gb[8] = {b0.x,b0.y,b0.z,b0.w, b1.x,b1.y,b1.z,b1.w};

        if (MODE == 0) {
            float4 xa0 = *(const float4*)(a1row + base);
            float4 xa1 = *(const float4*)(a1row + base + 4);
            float4 xb0 = *(const float4*)(a1row + base + 16);
            float4 xb1 = *(const float4*)(a1row + base + 20);
            const float gxa[8] = {xa0.x,xa0.y,xa0.z,xa0.w, xa1.x,xa1.y,xa1.z,xa1.w};
            const float gxb[8] = {xb0.x,xb0.y,xb0.z,xb0.w, xb1.x,xb1.y,xb1.z,xb1.w};
            #pragma unroll
            for (int i = 0; i < 8; ++i) {
                ga[i] = fmaf(coef, gxa[i], ga[i]);
                gb[i] = fmaf(coef, gxb[i], gb[i]);
            }
        } else {
            float4 sa0 = *(const float4*)(bnp + base);
            float4 sa1 = *(const float4*)(bnp + base + 4);
            float4 sb0 = *(const float4*)(bnp + base + 16);
            float4 sb1 = *(const float4*)(bnp + base + 20);
            float4 ca0 = *(const float4*)(bnp + 128 + base);
            float4 ca1 = *(const float4*)(bnp + 128 + base + 4);
            float4 cb0 = *(const float4*)(bnp + 128 + base + 16);
            float4 cb1 = *(const float4*)(bnp + 128 + base + 20);
            const float sca[8] = {sa0.x,sa0.y,sa0.z,sa0.w, sa1.x,sa1.y,sa1.z,sa1.w};
            const float scb[8] = {sb0.x,sb0.y,sb0.z,sb0.w, sb1.x,sb1.y,sb1.z,sb1.w};
            const float sha[8] = {ca0.x,ca0.y,ca0.z,ca0.w, ca1.x,ca1.y,ca1.z,ca1.w};
            const float shb[8] = {cb0.x,cb0.y,cb0.z,cb0.w, cb1.x,cb1.y,cb1.z,cb1.w};
            #pragma unroll
            for (int i = 0; i < 8; ++i) {
                float va = fmaf(ga[i], sca[i], sha[i]);
                float vb = fmaf(gb[i], scb[i], shb[i]);
                ga[i] = va > 0.0f ? va : 0.0f;
                gb[i] = vb > 0.0f ? vb : 0.0f;
            }
        }
        // ---- pack A fragment (ISA 7.12.2 16-bit A 16x32 layout) ----
        v16h af;
        #pragma unroll
        for (int i = 0; i < 8; ++i) {
            af[i]     = (_Float16)ga[i];
            af[8 + i] = (_Float16)gb[i];
        }
        // ---- 8 N-tiles from LDS, reuse A fragment ----
        #pragma unroll
        for (int nt = 0; nt < 8; ++nt) {
            int n0 = nt * 16 + l16;
            const _Float16* wrow = wsh + n0 * 136 + kt * 32 + half * 8;
            v8h w0 = *(const v8h*)(wrow);
            v8h w1 = *(const v8h*)(wrow + 16);
            v16h bf = __builtin_shufflevector(w0, w1,
                0,1,2,3,4,5,6,7,8,9,10,11,12,13,14,15);
            acc[nt] = __builtin_amdgcn_wmma_f32_16x16x32_f16(
                false, af, false, bf, (short)0, acc[nt], false, false);
        }
    }

    // ---- epilogue: C layout VGPR r -> M = r + half*8, N = l16 + nt*16 ----
    #pragma unroll
    for (int nt = 0; nt < 8; ++nt) {
        int col = nt * 16 + l16;
        float b = bias[col];
        #pragma unroll
        for (int r = 0; r < 8; ++r) {
            long row = rowBase + r + half * 8;
            if (row < NN) {
                float v = acc[nt][r] + b;
                if (MODE == 1) v = v > 0.0f ? v : 0.0f;
                out[row * CC + col] = v;
            }
        }
    }
}

// ---------------- column sums / sumsq for BN stats ----------------
__global__ __launch_bounds__(256) void colstats(const float* __restrict__ in,
                                                float* __restrict__ stats, int rows)
{
    int c = threadIdx.x & 127;
    int rh = threadIdx.x >> 7;
    long r0 = (long)blockIdx.x * 256 + rh;
    float s = 0.0f, ss = 0.0f;
    for (int i = 0; i < 128; ++i) {
        long r = r0 + i * 2;
        if (r < rows) {
            float v = in[r * CC + c];
            s += v; ss += v * v;
        }
    }
    __shared__ float sh[256];
    __shared__ float sh2[256];
    sh[threadIdx.x] = s; sh2[threadIdx.x] = ss;
    __syncthreads();
    if (threadIdx.x < 128) {
        atomicAdd(&stats[c],       sh[threadIdx.x] + sh[threadIdx.x + 128]);
        atomicAdd(&stats[128 + c], sh2[threadIdx.x] + sh2[threadIdx.x + 128]);
    }
}

// fold BN into per-column affine: y = x*a + c
__global__ void bnparam(const float* __restrict__ stats, const float* __restrict__ g,
                        const float* __restrict__ be, float* __restrict__ params, float invN)
{
    int c = threadIdx.x;
    float mean = stats[c] * invN;
    float var  = stats[128 + c] * invN - mean * mean;
    float inv  = rsqrtf(var + 1e-5f);
    float a    = g[c] * inv;
    params[c]        = a;
    params[128 + c]  = be[c] - mean * a;
}

// apply BN affine, optionally store node features, and pool per-graph (fused scatter)
__global__ __launch_bounds__(256) void apply_pool(
    const float* __restrict__ t, const float* __restrict__ params,
    const int* __restrict__ batch, float* __restrict__ xout,
    float* __restrict__ pool, int rows)
{
    long idx = (long)blockIdx.x * 256 + threadIdx.x;
    if (idx >= (long)rows * CC) return;
    long r = idx >> 7;
    int  c = idx & 127;
    float v = t[idx] * params[c] + params[128 + c];
    if (xout) xout[idx] = v;
    atomicAdd(&pool[(long)batch[r] * CC + c], v);
}

// ---------------- head ----------------
__global__ __launch_bounds__(256) void head1(const float* __restrict__ p1,
    const float* __restrict__ p2, const float* __restrict__ w,
    const float* __restrict__ b, float* __restrict__ h)
{
    __shared__ float hin[256];
    int g = blockIdx.x, t = threadIdx.x;
    hin[t] = (t < 128) ? p1[g * 128 + t] : p2[g * 128 + (t - 128)];
    __syncthreads();
    float acc = b[t];
    const float* wr = w + t * 256;
    #pragma unroll 8
    for (int i = 0; i < 256; ++i) acc += hin[i] * wr[i];
    h[g * 256 + t] = acc > 0.0f ? acc : 0.0f;
}

__global__ __launch_bounds__(256) void head2(const float* __restrict__ h,
    const float* __restrict__ w, const float* __restrict__ b, float* __restrict__ out)
{
    __shared__ float red[256];
    int g = blockIdx.x, t = threadIdx.x;
    red[t] = h[g * 256 + t] * w[t];
    __syncthreads();
    for (int s = 128; s > 0; s >>= 1) {
        if (t < s) red[t] += red[t + s];
        __syncthreads();
    }
    if (t == 0) out[g] = 1.0f / (1.0f + __expf(-(red[0] + b[0])));
}

// ---------------- orchestration ----------------
extern "C" void kernel_launch(void* const* d_in, const int* in_sizes, int n_in,
                              void* d_out, int out_size, void* d_ws, size_t ws_size,
                              hipStream_t stream) {
    const float* x     = (const float*)d_in[0];
    const int*   ei    = (const int*)  d_in[1];
    const float* ea    = (const float*)d_in[2];
    const int*   batch = (const int*)  d_in[3];
    const float* lw1   = (const float*)d_in[4];
    const float* lb1   = (const float*)d_in[5];
    const float* eps1  = (const float*)d_in[6];
    const float* W11   = (const float*)d_in[7];
    const float* b11   = (const float*)d_in[8];
    const float* g11   = (const float*)d_in[9];
    const float* be11  = (const float*)d_in[10];
    const float* W12   = (const float*)d_in[11];
    const float* b12   = (const float*)d_in[12];
    const float* g12   = (const float*)d_in[13];
    const float* be12  = (const float*)d_in[14];
    const float* lw2   = (const float*)d_in[15];
    const float* lb2   = (const float*)d_in[16];
    const float* eps2  = (const float*)d_in[17];
    const float* W21   = (const float*)d_in[18];
    const float* b21   = (const float*)d_in[19];
    const float* g21   = (const float*)d_in[20];
    const float* be21  = (const float*)d_in[21];
    const float* W22   = (const float*)d_in[22];
    const float* b22   = (const float*)d_in[23];
    const float* g22   = (const float*)d_in[24];
    const float* be22  = (const float*)d_in[25];
    const float* l1w   = (const float*)d_in[26];
    const float* l1b   = (const float*)d_in[27];
    const float* l2w   = (const float*)d_in[28];
    const float* l2b   = (const float*)d_in[29];

    const int* srci = ei;
    const int* dsti = ei + EE;

    char* base = (char*)d_ws;
    const size_t NB = (size_t)NN * CC * sizeof(float);      // 25.6 MB
    float*    agg    = (float*)(base);                      // reused as 't' after gemm1
    float*    h1     = (float*)(base + NB);
    float*    x1     = (float*)(base + 2 * NB);
    _Float16* w11h   = (_Float16*)(base + 3 * NB);
    _Float16* w12h   = w11h + CC * CC;
    _Float16* w21h   = w12h + CC * CC;
    _Float16* w22h   = w21h + CC * CC;
    float*    stats  = (float*)(w22h + CC * CC);            // 256
    float*    params1= stats + 256;                         // 256
    float*    params2= params1 + 256;                       // 256
    float*    pool1  = params2 + 256;                       // G*128
    float*    pool2  = pool1 + GG * CC;
    float*    hhead  = pool2 + GG * CC;                     // G*256

    const long NEL   = (long)NN * CC;
    const int  GZ    = (int)((NEL + 255) / 256);            // 25000
    const int  GMM   = (NN + 127) / 128;                    // 391 gemm blocks
    const int  GST   = (NN + 255) / 256;                    // 196 colstats blocks
    const int  GEDGE = (EE + EPB - 1) / EPB;                // 12500
    const float invN = 1.0f / (float)NN;

    // weights -> f16 (once per launch)
    wconv_f16<<<64, 256, 0, stream>>>(W11, w11h, CC * CC);
    wconv_f16<<<64, 256, 0, stream>>>(W12, w12h, CC * CC);
    wconv_f16<<<64, 256, 0, stream>>>(W21, w21h, CC * CC);
    wconv_f16<<<64, 256, 0, stream>>>(W22, w22h, CC * CC);

    zero_f32<<<(GG * CC * 2 + 255) / 256, 256, 0, stream>>>(pool1, (long)GG * CC * 2);

    // ---------- GIN layer 1 ----------
    zero_f32<<<GZ, 256, 0, stream>>>(agg, NEL);
    edge_msg<<<GEDGE, 256, 0, stream>>>(x, srci, dsti, ea, lw1, lb1, agg);
    gemm128<0><<<GMM, 256, 0, stream>>>(agg, x, eps1, nullptr, w11h, b11, h1);
    zero_f32<<<1, 256, 0, stream>>>(stats, 256);
    colstats<<<GST, 256, 0, stream>>>(h1, stats, NN);
    bnparam<<<1, 128, 0, stream>>>(stats, g11, be11, params1, invN);
    gemm128<1><<<GMM, 256, 0, stream>>>(h1, nullptr, nullptr, params1, w12h, b12, agg);
    zero_f32<<<1, 256, 0, stream>>>(stats, 256);
    colstats<<<GST, 256, 0, stream>>>(agg, stats, NN);
    bnparam<<<1, 128, 0, stream>>>(stats, g12, be12, params2, invN);
    apply_pool<<<GZ, 256, 0, stream>>>(agg, params2, batch, x1, pool1, NN);

    // ---------- GIN layer 2 ----------
    zero_f32<<<GZ, 256, 0, stream>>>(agg, NEL);
    edge_msg<<<GEDGE, 256, 0, stream>>>(x1, srci, dsti, ea, lw2, lb2, agg);
    gemm128<0><<<GMM, 256, 0, stream>>>(agg, x1, eps2, nullptr, w21h, b21, h1);
    zero_f32<<<1, 256, 0, stream>>>(stats, 256);
    colstats<<<GST, 256, 0, stream>>>(h1, stats, NN);
    bnparam<<<1, 128, 0, stream>>>(stats, g21, be21, params1, invN);
    gemm128<1><<<GMM, 256, 0, stream>>>(h1, nullptr, nullptr, params1, w22h, b22, agg);
    zero_f32<<<1, 256, 0, stream>>>(stats, 256);
    colstats<<<GST, 256, 0, stream>>>(agg, stats, NN);
    bnparam<<<1, 128, 0, stream>>>(stats, g22, be22, params2, invN);
    apply_pool<<<GZ, 256, 0, stream>>>(agg, params2, batch, nullptr, pool2, NN);

    // ---------- head ----------
    head1<<<GG, 256, 0, stream>>>(pool1, pool2, l1w, l1b, hhead);
    head2<<<GG, 256, 0, stream>>>(hhead, l2w, l2b, (float*)d_out);
}
